// Seq2Seq_46067819217210
// MI455X (gfx1250) — matrix operand order repaired and checked by reading, not compiled, not run on hardware
//
#include <hip/hip_runtime.h>

// ---------------------------------------------------------------------------
// Seq2Seq LSTM + attention for MI455X (gfx1250), wave32 + v_wmma_f32_16x16x32_f16.
// Encoder: 256 independent single-wave workgroups, each owns 16 batch rows and
// runs the full 112-step / 3-layer recurrence locally (h and c in LDS).
// Weights stream from L2 every step (anti-LICM barrier prevents the compiler
// from hoisting ~512 VGPRs of loop-invariant weight fragments and spilling).
// Attention energy GEMM hoisted out of the decoder loop (enc_out @ Wa_e.T once).
// ---------------------------------------------------------------------------

typedef _Float16 half8   __attribute__((ext_vector_type(8)));
typedef _Float16 half16  __attribute__((ext_vector_type(16)));
typedef float    float8  __attribute__((ext_vector_type(8)));

#define NB 4096   // batch
#define SL 112    // src len
#define HD 64     // hidden
#define TT 28     // decoder steps
#define GQ 256    // 4*H gates

__device__ __forceinline__ float sigm(float x){ return 1.0f/(1.0f+__expf(-x)); }
__device__ __forceinline__ float tanh_f(float x){
  float ax = fabsf(x);
  float e  = __expf(-2.0f*ax);
  float t  = (1.0f-e)/(1.0f+e);
  return x < 0.0f ? -t : t;
}

__device__ __forceinline__ half16 cat16(half8 lo, half8 hi){
  return __builtin_shufflevector(lo, hi, 0,1,2,3,4,5,6,7,8,9,10,11,12,13,14,15);
}

// A fragment (16x32 f16) from row-major [16][64] activations:
// lane L: row = L&15; halves 0..7 = K kf*32+hi*8.., halves 8..15 = K kf*32+16+hi*8..
__device__ __forceinline__ half16 load_afrag(const _Float16* base, int kf, int r15, int hi){
  const _Float16* p = base + r15*HD + kf*32 + hi*8;
  return cat16(*(const half8*)p, *(const half8*)(p+16));
}
// B fragment (32x16 f16) for gates = act @ W^T, W row-major [256][64]:
// lane L: col n = L&15; halves 0..7 = W[n][kf*32+hi*8..], halves 8..15 = W[n][kf*32+16+hi*8..].
__device__ __forceinline__ half16 load_bfrag(const _Float16* wrow, int kf, int hi){
  const _Float16* p = wrow + kf*32 + hi*8;
  return cat16(*(const half8*)p, *(const half8*)(p+16));
}
__device__ __forceinline__ float8 wmma_f16(half16 a, half16 b, float8 c){
  return __builtin_amdgcn_wmma_f32_16x16x32_f16(false, a, false, b, (short)0, c, false, false);
}

// acc[16 col-tiles of 16x16] += A(16x64) @ W(256x64)^T   (K=64 -> 2 k-fragments)
// B loads software-pipelined one tile ahead so L2 latency overlaps WMMA issue.
__device__ __forceinline__ void gemm_accum(const _Float16* Abase, const _Float16* W,
                                           float8 acc[16], int r15, int hi){
  const _Float16* wr = W + r15*HD;
#pragma unroll
  for (int kf = 0; kf < 2; ++kf){
    half16 a = load_afrag(Abase, kf, r15, hi);
    half16 b = load_bfrag(wr, kf, hi);
#pragma unroll
    for (int c = 0; c < 16; ++c){
      half16 bn = b;
      if (c < 15) bn = load_bfrag(wr + (c+1)*16*HD, kf, hi);
      acc[c] = wmma_f16(a, b, acc[c]);
      b = bn;
    }
  }
}

struct InPtrs { const float* p[31]; };

// ---------------------------------------------------------------------------
// Weight prep: f32 -> f16 packed [N=256][K=64] row-major, bias folding, splits.
// ---------------------------------------------------------------------------
__global__ void prep_kernel(InPtrs in,
    _Float16* __restrict__ encW, _Float16* __restrict__ decW,
    _Float16* __restrict__ dwih0r, _Float16* __restrict__ waH, _Float16* __restrict__ waE,
    float* __restrict__ wih0v, float* __restrict__ dwcol0,
    float* __restrict__ bsumE, float* __restrict__ bsumD, float* __restrict__ prev)
{
  int gid  = blockIdx.x*blockDim.x + threadIdx.x;
  int nthr = gridDim.x*blockDim.x;
  // enc: whh0, wih1, whh1, wih2, whh2 ; dec: same order
  const float* esrc[5] = { in.p[3],  in.p[6],  in.p[7],  in.p[10], in.p[11] };
  const float* dsrc[5] = { in.p[15], in.p[18], in.p[19], in.p[22], in.p[23] };
  for (int i = gid; i < GQ*HD; i += nthr){
#pragma unroll
    for (int s = 0; s < 5; ++s){
      encW[s*GQ*HD + i] = (_Float16)esrc[s][i];
      decW[s*GQ*HD + i] = (_Float16)dsrc[s][i];
    }
    int r = i / HD, c = i % HD;
    dwih0r[i] = (_Float16)in.p[14][r*65 + 1 + c];   // dec Wih0[:,1:65]
  }
  for (int i = gid; i < HD*HD; i += nthr){
    int r = i / HD, c = i % HD;
    waH[i] = (_Float16)in.p[26][r*128 + c];         // W_attn[:,0:64]  (hrep half)
    waE[i] = (_Float16)in.p[26][r*128 + 64 + c];    // W_attn[:,64:128] (enc half)
  }
  for (int i = gid; i < GQ; i += nthr){
    wih0v[i]  = in.p[2][i];                          // enc Wih0 [256,1]
    dwcol0[i] = in.p[14][i*65];                      // dec Wih0[:,0]
    bsumE[i]        = in.p[4][i]  + in.p[5][i];
    bsumE[GQ+i]     = in.p[8][i]  + in.p[9][i];
    bsumE[2*GQ+i]   = in.p[12][i] + in.p[13][i];
    bsumD[i]        = in.p[16][i] + in.p[17][i];
    bsumD[GQ+i]     = in.p[20][i] + in.p[21][i];
    bsumD[2*GQ+i]   = in.p[24][i] + in.p[25][i];
  }
  for (int i = gid; i < NB; i += nthr) prev[i] = in.p[1][i];
}

// ---------------------------------------------------------------------------
// Encoder: one wave per 16 batch rows, full 112-step 3-layer recurrence.
// h (f16) and c (f32) live in LDS; gate accumulators live in 128 VGPRs.
// ---------------------------------------------------------------------------
__global__ void __launch_bounds__(32) enc_kernel(
    const float* __restrict__ x, const float* __restrict__ wih0v,
    const float* __restrict__ bsumE, const _Float16* __restrict__ encW,
    _Float16* __restrict__ enc_out, _Float16* __restrict__ hdec, float* __restrict__ cdec)
{
  __shared__ alignas(16) _Float16 hbuf[3][16][HD];   // 6 KB
  __shared__ alignas(16) float    cbuf[3][16][HD];   // 12 KB
  int lane = threadIdx.x & 31;
  int r15 = lane & 15, hi = lane >> 4;
  int row0 = blockIdx.x * 16;

  for (int i = lane; i < 3*16*HD; i += 32){
    (&hbuf[0][0][0])[i] = (_Float16)0.0f;
    (&cbuf[0][0][0])[i] = 0.0f;
  }

  for (int t = 0; t < SL; ++t){
    // Anti-LICM: forbid hoisting loop-invariant weight/bias loads out of the
    // t-loop (the hoisted fragments would exceed the VGPR budget and spill).
    asm volatile("" ::: "memory");
#pragma unroll 1
    for (int l = 0; l < 3; ++l){
      float8 acc[16];
#pragma unroll
      for (int c = 0; c < 16; ++c)
#pragma unroll
        for (int e = 0; e < 8; ++e) acc[c][e] = 0.0f;

      const _Float16* hhW = encW + (size_t)(l==0 ? 0 : 2*l) * (GQ*HD);
      gemm_accum(&hbuf[l][0][0], hhW, acc, r15, hi);          // h_l(t-1) @ Whh^T
      if (l > 0){
        const _Float16* ihW = encW + (size_t)(2*l-1) * (GQ*HD);
        gemm_accum(&hbuf[l-1][0][0], ihW, acc, r15, hi);      // h_{l-1}(t) @ Wih^T
      }
      float xv[8];
      if (l == 0){
#pragma unroll
        for (int v = 0; v < 8; ++v) xv[v] = x[(row0 + hi*8 + v)*SL + t];
      }
#pragma unroll
      for (int c = 0; c < 4; ++c){
        int col = c*16 + r15;
        float bi = bsumE[l*GQ+col],     bf = bsumE[l*GQ+col+64];
        float bg = bsumE[l*GQ+col+128], bo = bsumE[l*GQ+col+192];
        float wi = 0.f, wf = 0.f, wg = 0.f, wo = 0.f;
        if (l == 0){ wi = wih0v[col]; wf = wih0v[col+64]; wg = wih0v[col+128]; wo = wih0v[col+192]; }
#pragma unroll
        for (int v = 0; v < 8; ++v){
          float gi = acc[c][v]+bi, gf = acc[c+4][v]+bf, gg = acc[c+8][v]+bg, go = acc[c+12][v]+bo;
          if (l == 0){ gi += xv[v]*wi; gf += xv[v]*wf; gg += xv[v]*wg; go += xv[v]*wo; }
          float I = sigm(gi), F = sigm(gf), G = tanh_f(gg), O = sigm(go);
          int m = v + 8*hi;
          float cn = F*cbuf[l][m][col] + I*G;
          cbuf[l][m][col] = cn;
          float hn = O*tanh_f(cn);
          hbuf[l][m][col] = (_Float16)hn;
          if (l == 2) enc_out[(size_t)(row0+m)*(SL*HD) + t*HD + col] = (_Float16)hn;
        }
      }
    }
  }
  // export final (h, c) as decoder initial state
#pragma unroll 1
  for (int l = 0; l < 3; ++l)
#pragma unroll
    for (int c = 0; c < 4; ++c)
#pragma unroll
      for (int v = 0; v < 8; ++v){
        int m = v + 8*hi, col = c*16 + r15;
        size_t off = (size_t)l*NB*HD + (size_t)(row0+m)*HD + col;
        hdec[off] = hbuf[l][m][col];
        cdec[off] = cbuf[l][m][col];
      }
}

// ---------------------------------------------------------------------------
// Generic [rows,64] @ [64,64]^T WMMA projection (used for ew precompute & hw).
// ---------------------------------------------------------------------------
__global__ void __launch_bounds__(128) proj64_kernel(
    const _Float16* __restrict__ A, const _Float16* __restrict__ W,
    const float* __restrict__ bias, _Float16* __restrict__ out)
{
  int wave = threadIdx.x >> 5, lane = threadIdx.x & 31;
  int r15 = lane & 15, hi = lane >> 4;
  size_t row0 = ((size_t)blockIdx.x*4 + wave)*16;
  float8 acc[4];
#pragma unroll
  for (int c = 0; c < 4; ++c)
#pragma unroll
    for (int e = 0; e < 8; ++e) acc[c][e] = 0.0f;
#pragma unroll
  for (int kf = 0; kf < 2; ++kf){
    half16 a = load_afrag(A + row0*HD, kf, r15, hi);
#pragma unroll
    for (int c = 0; c < 4; ++c){
      half16 b = load_bfrag(W + (c*16 + r15)*HD, kf, hi);
      acc[c] = wmma_f16(a, b, acc[c]);
    }
  }
#pragma unroll
  for (int c = 0; c < 4; ++c){
    int col = c*16 + r15;
    float bb = bias ? bias[col] : 0.0f;
#pragma unroll
    for (int v = 0; v < 8; ++v)
      out[(row0 + v + 8*hi)*HD + col] = (_Float16)(acc[c][v] + bb);
  }
}

// ---------------------------------------------------------------------------
// Attention: scores = v . tanh(ew[b,s] + hw[(b*S+s)%B]); softmax; weighted sum.
// One block per batch row b; everything L2-resident.
// ---------------------------------------------------------------------------
__global__ void __launch_bounds__(128) attn_kernel(
    const _Float16* __restrict__ ew, const _Float16* __restrict__ hw,
    const float* __restrict__ vvec, const _Float16* __restrict__ enc,
    _Float16* __restrict__ wbuf)
{
  __shared__ float sc[SL];
  __shared__ float sinv;
  int b = blockIdx.x, tid = threadIdx.x;
  if (tid < SL){
    int s = tid;
    int idx = (b*SL + s) & (NB - 1);
    const _Float16* ep = ew + (size_t)(b*SL + s)*HD;
    const _Float16* hp = hw + (size_t)idx*HD;
    float a = 0.0f;
#pragma unroll 4
    for (int j = 0; j < HD; ++j) a += vvec[j]*tanh_f((float)ep[j] + (float)hp[j]);
    sc[s] = a;
  }
  __syncthreads();
  if (tid == 0){
    float m = -1e30f;
    for (int s = 0; s < SL; ++s) m = fmaxf(m, sc[s]);
    float sum = 0.0f;
    for (int s = 0; s < SL; ++s){ float e = __expf(sc[s]-m); sc[s] = e; sum += e; }
    sinv = 1.0f/sum;
  }
  __syncthreads();
  if (tid < HD){
    float a = 0.0f;
    const _Float16* ep = enc + (size_t)b*SL*HD + tid;
    for (int s = 0; s < SL; ++s) a += sc[s]*(float)ep[s*HD];
    wbuf[(size_t)b*HD + tid] = (_Float16)(a*sinv);
  }
}

// ---------------------------------------------------------------------------
// Decoder step: 3-layer LSTM (WMMA gates) + fused output projection.
// ---------------------------------------------------------------------------
__global__ void __launch_bounds__(32) dec_step_kernel(int t,
    float* prev, const _Float16* __restrict__ wbuf,
    const float* __restrict__ dwcol0, const float* __restrict__ bsumD,
    const _Float16* __restrict__ decW, const _Float16* __restrict__ dwih0r,
    _Float16* hdec, float* cdec,
    const float* __restrict__ Wout, const float* __restrict__ bout,
    float* __restrict__ out)
{
  __shared__ alignas(16) _Float16 htop[16][HD];
  int lane = threadIdx.x & 31, r15 = lane & 15, hi = lane >> 4;
  int row0 = blockIdx.x * 16;
#pragma unroll 1
  for (int l = 0; l < 3; ++l){
    float8 acc[16];
#pragma unroll
    for (int c = 0; c < 16; ++c)
#pragma unroll
      for (int e = 0; e < 8; ++e) acc[c][e] = 0.0f;

    const _Float16* hhW = decW + (size_t)(l==0 ? 0 : 2*l)*(GQ*HD);
    gemm_accum(hdec + (size_t)l*NB*HD + (size_t)row0*HD, hhW, acc, r15, hi);   // old h_l
    const _Float16* ihW = (l==0) ? dwih0r : decW + (size_t)(2*l-1)*(GQ*HD);
    const _Float16* Ab  = (l==0) ? (wbuf + (size_t)row0*HD)
                                 : (hdec + (size_t)(l-1)*NB*HD + (size_t)row0*HD); // new h_{l-1}
    gemm_accum(Ab, ihW, acc, r15, hi);

    float pv[8];
    if (l == 0){
#pragma unroll
      for (int v = 0; v < 8; ++v) pv[v] = prev[row0 + hi*8 + v];
    }
#pragma unroll
    for (int c = 0; c < 4; ++c){
      int col = c*16 + r15;
      float bi = bsumD[l*GQ+col],     bf = bsumD[l*GQ+col+64];
      float bg = bsumD[l*GQ+col+128], bo = bsumD[l*GQ+col+192];
      float wi = 0.f, wf = 0.f, wg = 0.f, wo = 0.f;
      if (l == 0){ wi = dwcol0[col]; wf = dwcol0[col+64]; wg = dwcol0[col+128]; wo = dwcol0[col+192]; }
#pragma unroll
      for (int v = 0; v < 8; ++v){
        int m = v + 8*hi;
        size_t off = (size_t)l*NB*HD + (size_t)(row0+m)*HD + col;
        float gi = acc[c][v]+bi, gf = acc[c+4][v]+bf, gg = acc[c+8][v]+bg, go = acc[c+12][v]+bo;
        if (l == 0){ gi += pv[v]*wi; gf += pv[v]*wf; gg += pv[v]*wg; go += pv[v]*wo; }
        float I = sigm(gi), F = sigm(gf), G = tanh_f(gg), O = sigm(go);
        float co = cdec[off];
        float cn = F*co + I*G;
        cdec[off] = cn;
        float hn = O*tanh_f(cn);
        hdec[off] = (_Float16)hn;
        if (l == 2) htop[m][col] = (_Float16)hn;
      }
    }
  }
  __syncthreads();
  // xo = [h_top, weighted] @ W_out^T + b_out  (NF=1)
  if (lane < 16){
    int rb = row0 + lane;
    float a = bout[0];
#pragma unroll 8
    for (int j = 0; j < HD; ++j) a += (float)htop[lane][j]*Wout[j];
#pragma unroll 8
    for (int j = 0; j < HD; ++j) a += (float)wbuf[(size_t)rb*HD + j]*Wout[64+j];
    out[(size_t)t*NB + rb] = a;
    prev[rb] = a;
  }
}

// ---------------------------------------------------------------------------
extern "C" void kernel_launch(void* const* d_in, const int* in_sizes, int n_in,
                              void* d_out, int out_size, void* d_ws, size_t ws_size,
                              hipStream_t stream)
{
  (void)in_sizes; (void)out_size; (void)ws_size;
  char* ws = (char*)d_ws;
  size_t off = 0;
  auto take = [&](size_t bytes)->char*{
    char* r = ws + off; off = (off + bytes + 255) & ~(size_t)255; return r;
  };
  _Float16* encW   = (_Float16*)take((size_t)5*GQ*HD*2);
  _Float16* decW   = (_Float16*)take((size_t)5*GQ*HD*2);
  _Float16* dwih0r = (_Float16*)take((size_t)GQ*HD*2);
  _Float16* waH    = (_Float16*)take((size_t)HD*HD*2);
  _Float16* waE    = (_Float16*)take((size_t)HD*HD*2);
  float*    wih0v  = (float*)take((size_t)GQ*4);
  float*    dwcol0 = (float*)take((size_t)GQ*4);
  float*    bsumE  = (float*)take((size_t)3*GQ*4);
  float*    bsumD  = (float*)take((size_t)3*GQ*4);
  float*    prev   = (float*)take((size_t)NB*4);
  _Float16* hw     = (_Float16*)take((size_t)NB*HD*2);
  _Float16* wbuf   = (_Float16*)take((size_t)NB*HD*2);
  _Float16* hdec   = (_Float16*)take((size_t)3*NB*HD*2);
  float*    cdec   = (float*)take((size_t)3*NB*HD*4);
  _Float16* encout = (_Float16*)take((size_t)NB*SL*HD*2);
  _Float16* ew     = (_Float16*)take((size_t)NB*SL*HD*2);

  InPtrs P;
  for (int i = 0; i < 31; ++i) P.p[i] = (i < n_in) ? (const float*)d_in[i] : nullptr;

  prep_kernel<<<128, 256, 0, stream>>>(P, encW, decW, dwih0r, waH, waE,
                                       wih0v, dwcol0, bsumE, bsumD, prev);

  enc_kernel<<<NB/16, 32, 0, stream>>>((const float*)d_in[0], wih0v, bsumE, encW,
                                       encout, hdec, cdec);

  // Hoisted attention energy: ew = enc_out @ Wa_e^T + b_attn  ([B*S, 64])
  proj64_kernel<<<(NB*SL)/64, 128, 0, stream>>>(encout, waE, (const float*)d_in[27], ew);

  for (int t = 0; t < TT; ++t){
    // hw = h_top @ Wa_h^T  ([B, 64])
    proj64_kernel<<<NB/64, 128, 0, stream>>>(hdec + (size_t)2*NB*HD, waH, nullptr, hw);
    attn_kernel<<<NB, 128, 0, stream>>>(ew, hw, (const float*)d_in[28], encout, wbuf);
    dec_step_kernel<<<NB/16, 32, 0, stream>>>(t, prev, wbuf, dwcol0, bsumD, decW, dwih0r,
                                              hdec, cdec,
                                              (const float*)d_in[29], (const float*)d_in[30],
                                              (float*)d_out);
  }
}
